// LocalAggregation_13374528159872
// MI455X (gfx1250) — matrix operand order repaired and checked by reading, not compiled, not run on hardware
//
#include <hip/hip_runtime.h>
#include <hip/hip_bf16.h>

typedef __bf16 bf16;
typedef bf16  v16bf __attribute__((ext_vector_type(16)));
typedef bf16  v8bf  __attribute__((ext_vector_type(8)));
typedef bf16  v2bf  __attribute__((ext_vector_type(2)));
typedef float v8f   __attribute__((ext_vector_type(8)));

#define TILE_ST 72          // LDS row stride in bf16 elements (144B, 16B aligned)
#define WAVES_PER_BLK 8
#define EPS 1e-5f

union V16U { v16bf v; v8bf h[2]; };

// ---- A fragment: 16x32 bf16 per the CDNA5 ISA layout ----
// lane L<16: row=L, K = {k0..k0+7} U {k0+16..k0+23}
// lane L>=16: row=L-16, K = {k0+8..k0+15} U {k0+24..k0+31}
__device__ inline v16bf load_a_frag(const bf16* tile, int lane, int k0) {
    int row = lane & 15;
    int hi8 = (lane >> 4) << 3;
    const bf16* rp = tile + row * TILE_ST;
    V16U u;
    u.h[0] = *(const v8bf*)(rp + k0 + hi8);
    u.h[1] = *(const v8bf*)(rp + k0 + 16 + hi8);
    return u.v;
}

// ---- B fragment: pre-packed so each lane reads 16 contiguous bf16 ----
__device__ inline v16bf load_b_frag(const bf16* __restrict__ pk, int frag, int lane) {
    const bf16* p = pk + (((frag << 5) + lane) << 4);
    V16U u;
    u.h[0] = *(const v8bf*)p;
    u.h[1] = *(const v8bf*)(p + 8);
    return u.v;
}

// ---- gather 16 rows of feat (c_in=64) via idx, convert f32->bf16 into LDS tile ----
__device__ inline void gather_tile(const float* __restrict__ feat,
                                   const long long* __restrict__ idx,
                                   long long rowBase, int lane, bf16* tile) {
    long long my = 0;
    if (lane < 16) my = idx[rowBase + lane];
    int mlo = (int)(my & 0xffffffffLL);
    int mhi = (int)(my >> 32);
    for (int r = 0; r < 16; ++r) {
        int lo = __shfl(mlo, r, 32);
        int hi = __shfl(mhi, r, 32);
        long long src = ((long long)hi << 32) | (unsigned int)lo;
        const float2 f = *(const float2*)(feat + src * 64 + lane * 2);
        v2bf p;
        p.x = (bf16)f.x;
        p.y = (bf16)f.y;
        *(v2bf*)(tile + r * TILE_ST + lane * 2) = p;
    }
}

// ---- GEMM1: tile(16x64) @ W1(64x64) + b1 -> 4 C-tiles of 16x16 f32 ----
__device__ inline void wave_gemm1(const float* __restrict__ feat,
                                  const long long* __restrict__ idx,
                                  long long rowBase, int lane, bf16* tile,
                                  const bf16* __restrict__ W1p,
                                  const float* __restrict__ b1, v8f y1[4]) {
    gather_tile(feat, idx, rowBase, lane, tile);
    v16bf a0 = load_a_frag(tile, lane, 0);
    v16bf a1 = load_a_frag(tile, lane, 32);
    int col = lane & 15;
    for (int nt = 0; nt < 4; ++nt) {
        v8f c = {};
        c = __builtin_amdgcn_wmma_f32_16x16x32_bf16(false, a0, false,
                load_b_frag(W1p, nt, lane), (short)0, c, false, false);
        c = __builtin_amdgcn_wmma_f32_16x16x32_bf16(false, a1, false,
                load_b_frag(W1p, 4 + nt, lane), (short)0, c, false, false);
        float bias = b1[nt * 16 + col];
        for (int v = 0; v < 8; ++v) c[v] += bias;
        y1[nt] = c;
    }
}

// ---- write h1 = relu(bn1(y1)) back into the wave's LDS tile (C/D layout -> row-major) ----
__device__ inline void store_h1(const v8f y1[4], int lane, bf16* tile,
                                const float* __restrict__ scale1,
                                const float* __restrict__ shift1) {
    int col = lane & 15;
    int rbase = (lane >> 4) << 3;
    for (int nt = 0; nt < 4; ++nt) {
        int ch = nt * 16 + col;
        float sc = scale1[ch], sh = shift1[ch];
        for (int v = 0; v < 8; ++v) {
            float h = fmaxf(y1[nt][v] * sc + sh, 0.0f);
            tile[(rbase + v) * TILE_ST + ch] = (bf16)h;
        }
    }
}

// ---- GEMM2: h1(16x64) @ W2(64x128) + b2 -> 8 C-tiles ----
__device__ inline void wave_gemm2(int lane, const bf16* tile,
                                  const bf16* __restrict__ W2p,
                                  const float* __restrict__ b2, v8f y2[8]) {
    v16bf a0 = load_a_frag(tile, lane, 0);
    v16bf a1 = load_a_frag(tile, lane, 32);
    int col = lane & 15;
    for (int nt = 0; nt < 8; ++nt) {
        v8f c = {};
        c = __builtin_amdgcn_wmma_f32_16x16x32_bf16(false, a0, false,
                load_b_frag(W2p, nt, lane), (short)0, c, false, false);
        c = __builtin_amdgcn_wmma_f32_16x16x32_bf16(false, a1, false,
                load_b_frag(W2p, 8 + nt, lane), (short)0, c, false, false);
        float bias = b2[nt * 16 + col];
        for (int v = 0; v < 8; ++v) c[v] += bias;
        y2[nt] = c;
    }
}

// ---- per-channel sum/sumsq reduction of C-tiles into LDS ----
__device__ inline void accum_stats(const v8f* y, int ntiles, int lane,
                                   float* s_sum, float* s_sq) {
    int col = lane & 15;
    for (int nt = 0; nt < ntiles; ++nt) {
        float s = 0.0f, q = 0.0f;
        for (int v = 0; v < 8; ++v) {
            float x = y[nt][v];
            s += x;
            q += x * x;
        }
        s += __shfl_xor(s, 16, 32);
        q += __shfl_xor(q, 16, 32);
        if (lane < 16) {
            atomicAdd(&s_sum[nt * 16 + col], s);
            atomicAdd(&s_sq[nt * 16 + col], q);
        }
    }
}

// ======================= kernels =======================

__global__ void la_zero_stats(float* p, int count) {
    int i = blockIdx.x * blockDim.x + threadIdx.x;
    if (i < count) p[i] = 0.0f;
}

// Pack W1 [64x64] and W2 [64x128] (row-major [in][out] f32) into per-lane
// bf16 B-fragment order: frag(kt,nt): lane L -> col nt*16+(L&15),
// K = kt*32 + (L>=16?8:0) + (e<8 ? e : e+8), e=0..15.
__global__ void la_pack_weights(const float* __restrict__ W1,
                                const float* __restrict__ W2,
                                bf16* __restrict__ W1p, bf16* __restrict__ W2p) {
    int t = blockIdx.x * blockDim.x + threadIdx.x;
    if (t < 256) {                       // W1: 8 frags * 32 lanes
        int lane = t & 31, frag = t >> 5;
        int nt = frag & 3;
        int kt = frag >> 2;
        int n = nt * 16 + (lane & 15);
        int kb = kt * 32 + ((lane >> 4) << 3);
        bf16* dst = W1p + ((frag << 5) + lane) * 16;
        for (int e = 0; e < 16; ++e) dst[e] = (bf16)W1[(kb + (e < 8 ? e : e + 8)) * 64 + n];
    } else if (t < 256 + 512) {          // W2: 16 frags * 32 lanes
        int q = t - 256;
        int lane = q & 31, frag = q >> 5;
        int nt = frag & 7;
        int kt = frag >> 3;
        int n = nt * 16 + (lane & 15);
        int kb = kt * 32 + ((lane >> 4) << 3);
        bf16* dst = W2p + ((frag << 5) + lane) * 16;
        for (int e = 0; e < 16; ++e) dst[e] = (bf16)W2[(kb + (e < 8 ? e : e + 8)) * 128 + n];
    }
}

__global__ __launch_bounds__(256)
void la_gemm1_stats(const float* __restrict__ feat, const long long* __restrict__ idx,
                    const bf16* __restrict__ W1p, const float* __restrict__ b1,
                    float* __restrict__ gsum, float* __restrict__ gsq, int nStrips) {
    __shared__ __align__(16) unsigned short tileRaw[WAVES_PER_BLK * 16 * TILE_ST];
    __shared__ float s_sum[64];
    __shared__ float s_sq[64];
    int tid = threadIdx.x;
    if (tid < 64) { s_sum[tid] = 0.0f; s_sq[tid] = 0.0f; }
    __syncthreads();
    int wave = tid >> 5, lane = tid & 31;
    int strip = blockIdx.x * WAVES_PER_BLK + wave;
    if (strip < nStrips) {
        bf16* tile = ((bf16*)tileRaw) + wave * 16 * TILE_ST;
        v8f y1[4];
        wave_gemm1(feat, idx, (long long)strip * 16, lane, tile, W1p, b1, y1);
        accum_stats(y1, 4, lane, s_sum, s_sq);
    }
    __syncthreads();
    if (tid < 64) {
        atomicAdd(&gsum[tid], s_sum[tid]);
        atomicAdd(&gsq[tid], s_sq[tid]);
    }
}

__global__ void la_finalize(const float* __restrict__ sum, const float* __restrict__ sq,
                            const float* __restrict__ gamma, const float* __restrict__ beta,
                            float* __restrict__ scale, float* __restrict__ shift,
                            int C, float invN) {
    int c = blockIdx.x * blockDim.x + threadIdx.x;
    if (c < C) {
        float m = sum[c] * invN;
        float v = sq[c] * invN - m * m;
        float r = rsqrtf(v + EPS);
        float sc = gamma[c] * r;
        scale[c] = sc;
        shift[c] = beta[c] - m * sc;
    }
}

__global__ __launch_bounds__(256)
void la_gemm2_stats(const float* __restrict__ feat, const long long* __restrict__ idx,
                    const bf16* __restrict__ W1p, const float* __restrict__ b1,
                    const float* __restrict__ scale1, const float* __restrict__ shift1,
                    const bf16* __restrict__ W2p, const float* __restrict__ b2,
                    float* __restrict__ gsum, float* __restrict__ gsq, int nStrips) {
    __shared__ __align__(16) unsigned short tileRaw[WAVES_PER_BLK * 16 * TILE_ST];
    __shared__ float s_sum[128];
    __shared__ float s_sq[128];
    int tid = threadIdx.x;
    if (tid < 128) { s_sum[tid] = 0.0f; s_sq[tid] = 0.0f; }
    __syncthreads();
    int wave = tid >> 5, lane = tid & 31;
    int strip = blockIdx.x * WAVES_PER_BLK + wave;
    if (strip < nStrips) {
        bf16* tile = ((bf16*)tileRaw) + wave * 16 * TILE_ST;
        v8f y1[4];
        wave_gemm1(feat, idx, (long long)strip * 16, lane, tile, W1p, b1, y1);
        store_h1(y1, lane, tile, scale1, shift1);
        v8f y2[8];
        wave_gemm2(lane, tile, W2p, b2, y2);
        accum_stats(y2, 8, lane, s_sum, s_sq);
    }
    __syncthreads();
    if (tid < 128) {
        atomicAdd(&gsum[tid], s_sum[tid]);
        atomicAdd(&gsq[tid], s_sq[tid]);
    }
}

__global__ __launch_bounds__(256)
void la_final(const float* __restrict__ feat, const long long* __restrict__ idx,
              const bf16* __restrict__ W1p, const float* __restrict__ b1,
              const float* __restrict__ scale1, const float* __restrict__ shift1,
              const bf16* __restrict__ W2p, const float* __restrict__ b2,
              const float* __restrict__ scale2, const float* __restrict__ shift2,
              float* __restrict__ out, int nStrips) {
    __shared__ __align__(16) unsigned short tileRaw[WAVES_PER_BLK * 16 * TILE_ST];
    int tid = threadIdx.x;
    int wave = tid >> 5, lane = tid & 31;
    int strip = blockIdx.x * WAVES_PER_BLK + wave;
    if (strip >= nStrips) return;
    bf16* tile = ((bf16*)tileRaw) + wave * 16 * TILE_ST;
    v8f y1[4];
    wave_gemm1(feat, idx, (long long)strip * 16, lane, tile, W1p, b1, y1);
    store_h1(y1, lane, tile, scale1, shift1);
    v8f y2[8];
    wave_gemm2(lane, tile, W2p, b2, y2);
    int col = lane & 15;
    for (int nt = 0; nt < 8; ++nt) {
        int ch = nt * 16 + col;
        float sc = scale2[ch], sh = shift2[ch];
        float m = -3.4e38f;
        for (int v = 0; v < 8; ++v)
            m = fmaxf(m, fmaxf(y2[nt][v] * sc + sh, 0.0f));
        m = fmaxf(m, __shfl_xor(m, 16, 32));
        if (lane < 16) out[(long long)strip * 128 + ch] = m;
    }
}

// ======================= host launcher =======================

extern "C" void kernel_launch(void* const* d_in, const int* in_sizes, int n_in,
                              void* d_out, int out_size, void* d_ws, size_t ws_size,
                              hipStream_t stream) {
    const float*     feat = (const float*)d_in[0];
    const long long* idx  = (const long long*)d_in[1];
    const float* W1  = (const float*)d_in[2];
    const float* b1  = (const float*)d_in[3];
    const float* g1  = (const float*)d_in[4];
    const float* be1 = (const float*)d_in[5];
    const float* W2  = (const float*)d_in[6];
    const float* b2  = (const float*)d_in[7];
    const float* g2  = (const float*)d_in[8];
    const float* be2 = (const float*)d_in[9];
    float* out = (float*)d_out;

    int nk = in_sizes[1];          // n * k (k == 16)
    int nStrips = nk / 16;         // one 16-row strip per output point
    float invN = 1.0f / (float)nk;

    // workspace layout (~28 KB total)
    char* ws = (char*)d_ws;
    bf16* W1p = (bf16*)ws;                    //  8192 B
    bf16* W2p = (bf16*)(ws + 8192);           // 16384 B
    float* st = (float*)(ws + 24576);
    float* sum1 = st;            // 64
    float* sq1  = st + 64;       // 64
    float* sum2 = st + 128;      // 128
    float* sq2  = st + 256;      // 128
    float* scale1 = st + 384;    // 64
    float* shift1 = st + 448;    // 64
    float* scale2 = st + 512;    // 128
    float* shift2 = st + 640;    // 128

    int blocks = (nStrips + WAVES_PER_BLK - 1) / WAVES_PER_BLK;

    la_zero_stats<<<1, 384, 0, stream>>>(st, 384);
    la_pack_weights<<<3, 256, 0, stream>>>(W1, W2, W1p, W2p);
    la_gemm1_stats<<<blocks, 256, 0, stream>>>(feat, idx, W1p, b1, sum1, sq1, nStrips);
    la_finalize<<<1, 64, 0, stream>>>(sum1, sq1, g1, be1, scale1, shift1, 64, invN);
    la_gemm2_stats<<<blocks, 256, 0, stream>>>(feat, idx, W1p, b1, scale1, shift1,
                                               W2p, b2, sum2, sq2, nStrips);
    la_finalize<<<1, 128, 0, stream>>>(sum2, sq2, g2, be2, scale2, shift2, 128, invN);
    la_final<<<blocks, 256, 0, stream>>>(feat, idx, W1p, b1, scale1, shift1,
                                         W2p, b2, scale2, shift2, out, nStrips);
}
